// GCNLayer_83562883711391
// MI455X (gfx1250) — compile-verified, matrix operand order checked by reading, not run on hardware
//
#include <hip/hip_runtime.h>

#define B_DIM 8
#define N_DIM 2048
#define F_DIM 256   // IN_DIM == OUT_DIM == 256

typedef __attribute__((ext_vector_type(16))) __bf16 bf16x16;
typedef __attribute__((ext_vector_type(8)))  float  f32x8;

union BfragU { bf16x16 v; uint4 q[2]; };

// 16-bit WMMA operand K index for element e (0..15) in lane-half h (0/1):
// VGPR pairs hold K = {8h..8h+7} then {16+8h..23+8h}  (ISA 7.12.2)
__device__ __forceinline__ int kmap(int e, int h) {
    return e + 8 * h + ((e >= 8) ? 8 : 0);
}

// Build bf16 A-fragment from an f32 LDS row (two contiguous 8-float runs)
__device__ __forceinline__ bf16x16 make_afrag_f32(const float* ar, int h) {
    bf16x16 a;
#pragma unroll
    for (int e = 0; e < 16; ++e) a[e] = (__bf16)ar[kmap(e, h)];
    return a;
}

// Load 4 B-fragments (one 32-K chunk, 4 N-tiles) from transposed bf16 matrix
__device__ __forceinline__ void load_bfrags(const __bf16* xb_base, int jc, BfragU bu[4]) {
    const __bf16* xb = xb_base + jc;
#pragma unroll
    for (int t = 0; t < 4; ++t) {
        const __bf16* p = xb + (size_t)(16 * t) * N_DIM;
        bu[t].q[0] = *(const uint4*)(p);
        bu[t].q[1] = *(const uint4*)(p + 16);
    }
}

// Async DMA: 16B global -> LDS per lane, tracked by ASYNCcnt (no VGPR data path).
// INST_OFFSET is added to BOTH global and LDS addresses (ISA 10.x async ops).
__device__ __forceinline__ void async_g2l_b128(unsigned lds_addr, const void* gptr) {
    asm volatile("global_load_async_to_lds_b128 %0, %1, off"
                 :: "v"(lds_addr), "v"(gptr) : "memory");
}
__device__ __forceinline__ void async_g2l_b128_off16(unsigned lds_addr, const void* gptr) {
    asm volatile("global_load_async_to_lds_b128 %0, %1, off offset:16"
                 :: "v"(lds_addr), "v"(gptr) : "memory");
}
__device__ __forceinline__ void wait_async0() {
    asm volatile("s_wait_asynccnt 0x0" ::: "memory");
}

// ---------------- kernel 1: d[b,n] = rsqrt(1 + rowsum(adj)) ----------------
__global__ __launch_bounds__(256) void deg_kernel(const float* __restrict__ adj,
                                                  float* __restrict__ ddeg) {
    const int w = threadIdx.x >> 5, lane = threadIdx.x & 31;
    const int row = blockIdx.x * 8 + w;               // 0..B*N-1
    const float* r = adj + (size_t)row * N_DIM;
    float s = 0.f;
    for (int j = lane * 4; j < N_DIM; j += 32 * 4) {
        float4 v = *(const float4*)(r + j);
        s += v.x + v.y + v.z + v.w;
    }
#pragma unroll
    for (int off = 16; off >= 1; off >>= 1) s += __shfl_xor(s, off, 32);
    if (lane == 0) ddeg[row] = rsqrtf(s + 1.0f);
}

// ---------------- kernel 2: W (f32, [out][in]) -> bf16 ----------------
__global__ __launch_bounds__(256) void wbf_kernel(const float* __restrict__ W,
                                                  __bf16* __restrict__ Wbf) {
    int i = (blockIdx.x * 256 + threadIdx.x) * 4;     // 64 blocks cover 65536
    float4 v = *(const float4*)(W + i);
    Wbf[i + 0] = (__bf16)v.x;
    Wbf[i + 1] = (__bf16)v.y;
    Wbf[i + 2] = (__bf16)v.z;
    Wbf[i + 3] = (__bf16)v.w;
}

// -------- kernel 3: Xst[b][f][j] = bf16(d[b,j] * X[b,j,f])  (transposed) ----
__global__ __launch_bounds__(256) void xst_kernel(const float* __restrict__ X,
                                                  const float* __restrict__ ddeg,
                                                  __bf16* __restrict__ Xst) {
    __shared__ __bf16 tbuf[F_DIM * 34];
    const int b = blockIdx.y;
    const int j0 = blockIdx.x * 32;
    const int tid = threadIdx.x;                      // 0..255 == f
#pragma unroll 4
    for (int i = 0; i < 32; ++i) {
        int j = j0 + i;
        float val = ddeg[b * N_DIM + j] * X[((size_t)(b * N_DIM + j)) * F_DIM + tid];
        tbuf[tid * 34 + i] = (__bf16)val;
    }
    __syncthreads();
#pragma unroll 4
    for (int it = 0; it < 32; ++it) {
        int idx = it * 256 + tid;
        int f = idx >> 5, jj = idx & 31;
        Xst[((size_t)(b * F_DIM + f)) * N_DIM + j0 + jj] = tbuf[f * 34 + jj];
    }
}

// ---------------- kernel 4: fused GCN layer ----------------
// 4 waves/block, 32-row tile x 256 cols. Wave w owns 64 feature/output columns,
// two 16-row M tiles. adj tiles stream HBM->LDS via async DMA (ASYNCcnt, no
// VGPR staging), B fragments double-buffered in registers.
__global__ __launch_bounds__(128) void gcn_main(const float* __restrict__ adj,
                                                const float* __restrict__ X,
                                                const float* __restrict__ bias,
                                                const float* __restrict__ ddeg,
                                                const __bf16* __restrict__ Xst,
                                                const __bf16* __restrict__ Wbf,
                                                float* __restrict__ out) {
    __shared__ alignas(16) float  lds_adj[2][32 * 32];   // double-buffered adj tile (4KB x2)
    __shared__ alignas(16) __bf16 Hl[32 * 264];          // H tile, 16B-aligned padded stride

    const int b    = blockIdx.y;
    const int i0   = blockIdx.x * 32;
    const int tid  = threadIdx.x;
    const int w    = tid >> 5;
    const int lane = tid & 31;
    const int n    = lane & 15;      // N index within 16x16 tile
    const int h    = lane >> 4;      // lane half
    const int fbase = w * 64;

    const float*  adjb = adj + ((size_t)b * N_DIM + i0) * N_DIM;
    const __bf16* xb   = Xst + ((size_t)(b * F_DIM + fbase + n)) * N_DIM + 8 * h;

    const int rr  = tid >> 2;              // 0..31 (row, 4 threads/row)
    const int cc0 = (tid & 3) * 8;         // 0,8,16,24 (two b128 copies per thread)

    // per-thread LDS byte addresses for the two tile buffers
    const unsigned ldsA0 = (unsigned)(uintptr_t)&lds_adj[0][rr * 32 + cc0];
    const unsigned ldsA1 = (unsigned)(uintptr_t)&lds_adj[1][rr * 32 + cc0];

    f32x8  acc[2][4] = {};
    BfragU bufs[2][4];

    // ---- prologue: DMA tile 0 -> LDS[0], B chunk 0 -> bufs[0]
    {
        const float* gp = adjb + (size_t)rr * N_DIM + cc0;
        async_g2l_b128(ldsA0, gp);
        async_g2l_b128_off16(ldsA0, gp);
        load_bfrags(xb, 0, bufs[0]);
        wait_async0();
        __syncthreads();
    }

    // ---- main K loop, 2 chunks (64 K) per iteration so reg buffers swap roles
    for (int jc = 0; jc < N_DIM; jc += 64) {
#pragma unroll
        for (int p = 0; p < 2; ++p) {
            const int c = jc + 32 * p;          // current chunk; LDS buf == p
            const bool more = (c + 32) < N_DIM;

            // 1) kick off next adj tile DMA + next B-fragment loads
            if (more) {
                const float* gp = adjb + (size_t)rr * N_DIM + (c + 32) + cc0;
                unsigned la = p ? ldsA0 : ldsA1;
                async_g2l_b128(la, gp);
                async_g2l_b128_off16(la, gp);
                load_bfrags(xb, c + 32, bufs[p ^ 1]);
            }

            // 2) A fragments (two M tiles) from LDS, f32 -> bf16
            bf16x16 a0 = make_afrag_f32(&lds_adj[p][n * 32], h);
            bf16x16 a1 = make_afrag_f32(&lds_adj[p][(16 + n) * 32], h);

            // 3) 8 WMMAs (each B fragment reused across both M tiles)
#pragma unroll
            for (int t = 0; t < 4; ++t) {
                acc[0][t] = __builtin_amdgcn_wmma_f32_16x16x32_bf16(
                    false, a0, false, bufs[p][t].v, (short)0, acc[0][t], false, false);
                acc[1][t] = __builtin_amdgcn_wmma_f32_16x16x32_bf16(
                    false, a1, false, bufs[p][t].v, (short)0, acc[1][t], false, false);
            }

            // 4) make next tile visible: drain this wave's DMA, then barrier
            wait_async0();
            __syncthreads();
        }
    }

    // ---- epilogue A: H = d_i * (acc + d_i * X[i,f])  (diagonal of A+I folded in)
    float dvv[2][8];
#pragma unroll
    for (int m = 0; m < 2; ++m)
#pragma unroll
        for (int r = 0; r < 8; ++r)
            dvv[m][r] = ddeg[b * N_DIM + i0 + m * 16 + r + 8 * h];

#pragma unroll
    for (int m = 0; m < 2; ++m)
#pragma unroll
        for (int t = 0; t < 4; ++t) {
            int f = fbase + 16 * t + n;
#pragma unroll
            for (int r = 0; r < 8; ++r) {
                int M = m * 16 + r + 8 * h;
                float x  = X[((size_t)(b * N_DIM + i0 + M)) * F_DIM + f];
                float hv = dvv[m][r] * (acc[m][t][r] + dvv[m][r] * x);
                Hl[M * 264 + f] = (__bf16)hv;
            }
        }
    __syncthreads();

    // ---- epilogue B: out = relu(H @ W^T + bias); wave w owns o in [fbase, fbase+64)
    f32x8 acc2[2][4] = {};
#pragma unroll
    for (int fc = 0; fc < F_DIM; fc += 32) {
        BfragU a20, a21;        // A fragments straight from bf16 LDS (16B runs)
        {
            const __bf16* hr0 = &Hl[n * 264 + fc];
            const __bf16* hr1 = &Hl[(16 + n) * 264 + fc];
            a20.q[0] = *(const uint4*)(hr0 + 8 * h);
            a20.q[1] = *(const uint4*)(hr0 + 16 + 8 * h);
            a21.q[0] = *(const uint4*)(hr1 + 8 * h);
            a21.q[1] = *(const uint4*)(hr1 + 16 + 8 * h);
        }
        const __bf16* wb = Wbf + (size_t)(fbase + n) * F_DIM + fc + 8 * h;
#pragma unroll
        for (int t = 0; t < 4; ++t) {
            BfragU bu;
            const __bf16* pp = wb + (size_t)(16 * t) * F_DIM;
            bu.q[0] = *(const uint4*)pp;
            bu.q[1] = *(const uint4*)(pp + 16);
            acc2[0][t] = __builtin_amdgcn_wmma_f32_16x16x32_bf16(
                false, a20.v, false, bu.v, (short)0, acc2[0][t], false, false);
            acc2[1][t] = __builtin_amdgcn_wmma_f32_16x16x32_bf16(
                false, a21.v, false, bu.v, (short)0, acc2[1][t], false, false);
        }
    }

#pragma unroll
    for (int m = 0; m < 2; ++m)
#pragma unroll
        for (int t = 0; t < 4; ++t) {
            int o = fbase + 16 * t + n;
            float bo = bias[o];
#pragma unroll
            for (int r = 0; r < 8; ++r) {
                int M = m * 16 + r + 8 * h;
                float v = acc2[m][t][r] + bo;
                out[((size_t)(b * N_DIM + i0 + M)) * F_DIM + o] = v > 0.f ? v : 0.f;
            }
        }
}

extern "C" void kernel_launch(void* const* d_in, const int* in_sizes, int n_in,
                              void* d_out, int out_size, void* d_ws, size_t ws_size,
                              hipStream_t stream) {
    (void)in_sizes; (void)n_in; (void)out_size; (void)ws_size;
    const float* X    = (const float*)d_in[0];   // [8,2048,256]
    const float* adj  = (const float*)d_in[1];   // [8,2048,2048]
    const float* W    = (const float*)d_in[2];   // [256,256]
    const float* bias = (const float*)d_in[3];   // [256]
    float* out = (float*)d_out;

    char* ws = (char*)d_ws;
    float*  ddeg = (float*)ws;                       //   64 KB : d[b,n]
    __bf16* Wbf  = (__bf16*)(ws + 64 * 1024);        //  128 KB : bf16 W
    __bf16* Xst  = (__bf16*)(ws + 256 * 1024);       //    8 MB : bf16 d_j*X, [b][f][j]

    deg_kernel<<<2048, 256, 0, stream>>>(adj, ddeg);
    wbf_kernel<<<64, 256, 0, stream>>>(W, Wbf);
    xst_kernel<<<dim3(64, B_DIM), 256, 0, stream>>>(X, ddeg, Xst);
    gcn_main<<<dim3(N_DIM / 32, B_DIM), 128, 0, stream>>>(adj, X, bias, ddeg, Xst, Wbf, out);
}
